// Detector_73804718014475
// MI455X (gfx1250) — matrix operand order repaired and checked by reading, not compiled, not run on hardware
//
#include <hip/hip_runtime.h>
#include <hip/hip_bf16.h>
#include <math.h>

typedef __attribute__((ext_vector_type(16))) _Float16 v16h;
typedef __attribute__((ext_vector_type(8)))  float    v8f;

#define B_   16
#define L_   24
#define T_   400
#define D_   1024
#define E_   512
#define RIN_ 12288
#define RH_  128
#define NC_  2
#define TEMP_  1.75f
#define ALPHA_ 0.325f
#define TSPLIT_ 5          // T chunks per (b,l)
#define CHUNK_T_ 80        // 5 tiles of 16

#define WMMA_F16(a, b, c) \
  __builtin_amdgcn_wmma_f32_16x16x32_f16(false, (a), false, (b), (short)0, (c), false, false)

// ---------------------------------------------------------------------------
// K0: convert fp32 weights -> f16, pre-swizzled into WMMA B-fragment layout.
// B (KxN, 16-bit): lane = (n&15) + 16*((k32>>3)&1), half = 8*(k32>>4)+(k32&7)
// Tile = 32(K) x 16(N) = 512 halves = 1KB, stored lane-major (16 halves/lane).
// ---------------------------------------------------------------------------
__global__ __launch_bounds__(256)
void convert_weights(const float* __restrict__ We1, const float* __restrict__ We2,
                     _Float16* __restrict__ We1h, _Float16* __restrict__ We2h) {
  const int total1 = L_ * D_ * E_;   // 12,582,912
  const int total2 = L_ * E_ * E_;   //  6,291,456
  for (int idx = blockIdx.x * blockDim.x + threadIdx.x; idx < total1 + total2;
       idx += gridDim.x * blockDim.x) {
    if (idx < total1) {
      int l = idx / (D_ * E_);
      int r = idx - l * (D_ * E_);
      int k = r >> 9;            // / E_
      int n = r & (E_ - 1);
      int k32 = k & 31;
      int ln  = (n & 15) + (((k32 >> 3) & 1) << 4);
      int hi  = ((k32 >> 4) << 3) + (k32 & 7);
      size_t tile = ((size_t)l * 32 + (k >> 5)) * 32 + (n >> 4);
      We1h[tile * 512 + ln * 16 + hi] = (_Float16)We1[idx];
    } else {
      int j = idx - total1;
      int l = j / (E_ * E_);
      int r = j - l * (E_ * E_);
      int k = r >> 9;
      int n = r & (E_ - 1);
      int k32 = k & 31;
      int ln  = (n & 15) + (((k32 >> 3) & 1) << 4);
      int hi  = ((k32 >> 4) << 3) + (k32 & 7);
      size_t tile = ((size_t)l * 16 + (k >> 5)) * 32 + (n >> 4);
      We2h[tile * 512 + ln * 16 + hi] = (_Float16)We2[j];
    }
  }
}

// ---------------------------------------------------------------------------
// K1: fused expert MLP (GEMM1+ReLU+GEMM2) + online-softmax attention pooling
// over a chunk of 80 timesteps. One block per (b,l,chunk). 8 waves; wave w
// owns output columns [64w, 64w+64). Emits partial (acc, m, s) to workspace.
// ---------------------------------------------------------------------------
__global__ __launch_bounds__(256)
void expert_pool_kernel(const float* __restrict__ h,
                        const _Float16* __restrict__ We1h,
                        const _Float16* __restrict__ We2h,
                        const float* __restrict__ be1,
                        const float* __restrict__ be2,
                        const float* __restrict__ wa,
                        const float* __restrict__ ba,
                        float* __restrict__ part_acc,
                        float* __restrict__ part_ms) {
  __shared__ __attribute__((aligned(32))) _Float16 aT[32 * 512]; // 32KB A-frags, GEMM1
  __shared__ __attribute__((aligned(32))) _Float16 xT[16 * 512]; // 16KB A-frags, GEMM2
  __shared__ float scoreBuf[8 * 16];
  __shared__ float wts[16];
  __shared__ float sCorr, sRunMax, sRunSum;

  const int tid   = threadIdx.x;
  const int lane  = tid & 31;
  const int w     = tid >> 5;
  const int blk   = blockIdx.x;
  const int c     = blk % TSPLIT_;
  const int bl    = blk / TSPLIT_;
  const int b     = bl / L_;
  const int l     = bl - b * L_;
  const int nloc  = lane & 15;
  const int hhalf = lane >> 4;   // C-frag M offset: 0 or 8

  float waj[4], be1j[4], be2j[4];
  #pragma unroll
  for (int j = 0; j < 4; ++j) {
    int n = w * 64 + j * 16 + nloc;
    waj[j]  = wa[n];
    be1j[j] = be1[l * E_ + n];
    be2j[j] = be2[l * E_ + n];
  }

  float pool[4] = {0.f, 0.f, 0.f, 0.f};
  if (tid == 0) { sRunMax = -3.0e38f; sRunSum = 0.f; }

  const float*    hBase  = h + (size_t)(b * L_ + l) * T_ * D_;
  const _Float16* w1Base = We1h + (size_t)l * 32 * 32 * 512;
  const _Float16* w2Base = We2h + (size_t)l * 16 * 32 * 512;

  for (int t0 = c * CHUNK_T_; t0 < (c + 1) * CHUNK_T_; t0 += 16) {
    __syncthreads();                                   // B1
    // ---- stage h tile [16,1024] fp32 -> f16 swizzled A-fragments ----
    const float* hT = hBase + (size_t)t0 * D_;
    for (int i = tid; i < 16 * 1024; i += 256) {
      int tl = i >> 10;
      int d  = i & 1023;
      float v = hT[(size_t)tl * D_ + d];
      int k32 = d & 31;
      int ln  = (tl & 15) + (((k32 >> 3) & 1) << 4);
      int vv  = ((k32 & 0x10) >> 2) | ((k32 >> 1) & 3);
      int hi  = 2 * vv + (k32 & 1);
      aT[(d >> 5) * 512 + ln * 16 + hi] = (_Float16)v;
    }
    __syncthreads();                                   // B2

    // ---- GEMM1: [16,1024] x [1024,64(per wave)], software pipelined ----
    v8f acc1[4] = {};
    {
      v16h a = *(const v16h*)(&aT[lane * 16]);
      const _Float16* bt0 = w1Base + (size_t)(w * 4) * 512 + lane * 16;
      v16h b0 = *(const v16h*)(bt0);
      v16h b1 = *(const v16h*)(bt0 + 512);
      v16h b2 = *(const v16h*)(bt0 + 1024);
      v16h b3 = *(const v16h*)(bt0 + 1536);
      #pragma unroll 4
      for (int kt = 0; kt < 32; ++kt) {
        int ktn = (kt + 1) & 31;
        v16h an = *(const v16h*)(&aT[ktn * 512 + lane * 16]);
        const _Float16* btn = w1Base + ((size_t)ktn * 32 + w * 4) * 512 + lane * 16;
        v16h n0 = *(const v16h*)(btn);
        v16h n1 = *(const v16h*)(btn + 512);
        v16h n2 = *(const v16h*)(btn + 1024);
        v16h n3 = *(const v16h*)(btn + 1536);
        acc1[0] = WMMA_F16(a, b0, acc1[0]);
        acc1[1] = WMMA_F16(a, b1, acc1[1]);
        acc1[2] = WMMA_F16(a, b2, acc1[2]);
        acc1[3] = WMMA_F16(a, b3, acc1[3]);
        a = an; b0 = n0; b1 = n1; b2 = n2; b3 = n3;
      }
    }
    // ---- bias + ReLU, scatter into A-fragment layout for GEMM2 ----
    #pragma unroll
    for (int j = 0; j < 4; ++j) {
      int kk  = w * 64 + j * 16 + nloc;       // GEMM2 K index
      int k32 = kk & 31;
      int kt2 = kk >> 5;
      int vv  = ((k32 & 0x10) >> 2) | ((k32 >> 1) & 3);
      int hi  = 2 * vv + (k32 & 1);
      int lnb = ((k32 >> 3) & 1) << 4;
      #pragma unroll
      for (int r = 0; r < 8; ++r) {
        int m = r + 8 * hhalf;
        float v = acc1[j][r] + be1j[j];
        v = v > 0.f ? v : 0.f;
        xT[kt2 * 512 + (m + lnb) * 16 + hi] = (_Float16)v;
      }
    }
    __syncthreads();                                   // B3

    // ---- GEMM2: [16,512] x [512,64(per wave)], software pipelined ----
    v8f acc2[4] = {};
    {
      v16h a = *(const v16h*)(&xT[lane * 16]);
      const _Float16* bt0 = w2Base + (size_t)(w * 4) * 512 + lane * 16;
      v16h b0 = *(const v16h*)(bt0);
      v16h b1 = *(const v16h*)(bt0 + 512);
      v16h b2 = *(const v16h*)(bt0 + 1024);
      v16h b3 = *(const v16h*)(bt0 + 1536);
      #pragma unroll 4
      for (int kt = 0; kt < 16; ++kt) {
        int ktn = (kt + 1) & 15;
        v16h an = *(const v16h*)(&xT[ktn * 512 + lane * 16]);
        const _Float16* btn = w2Base + ((size_t)ktn * 32 + w * 4) * 512 + lane * 16;
        v16h n0 = *(const v16h*)(btn);
        v16h n1 = *(const v16h*)(btn + 512);
        v16h n2 = *(const v16h*)(btn + 1024);
        v16h n3 = *(const v16h*)(btn + 1536);
        acc2[0] = WMMA_F16(a, b0, acc2[0]);
        acc2[1] = WMMA_F16(a, b1, acc2[1]);
        acc2[2] = WMMA_F16(a, b2, acc2[2]);
        acc2[3] = WMMA_F16(a, b3, acc2[3]);
        a = an; b0 = n0; b1 = n1; b2 = n2; b3 = n3;
      }
    }
    #pragma unroll
    for (int j = 0; j < 4; ++j)
      #pragma unroll
      for (int r = 0; r < 8; ++r) acc2[j][r] += be2j[j];

    // ---- attention scores: s[t] = x2[t,:].wa (+ba later) ----
    float s[8];
    #pragma unroll
    for (int r = 0; r < 8; ++r) {
      float acc = 0.f;
      #pragma unroll
      for (int j = 0; j < 4; ++j) acc += acc2[j][r] * waj[j];
      s[r] = acc;
    }
    #pragma unroll
    for (int off = 1; off < 16; off <<= 1)
      #pragma unroll
      for (int r = 0; r < 8; ++r) s[r] += __shfl_xor(s[r], off, 32);
    if (nloc == 0) {   // lanes 0 and 16 hold reduced halves
      #pragma unroll
      for (int r = 0; r < 8; ++r) scoreBuf[w * 16 + hhalf * 8 + r] = s[r];
    }
    __syncthreads();                                   // B4

    // ---- online softmax (lanes 0..15 of wave 0) ----
    if (tid < 16) {
      float sc = ba[0];
      #pragma unroll
      for (int ww = 0; ww < 8; ++ww) sc += scoreBuf[ww * 16 + tid];
      float mx = sc;
      for (int off = 1; off < 16; off <<= 1) mx = fmaxf(mx, __shfl_xor(mx, off, 32));
      float rm = sRunMax;
      float nm = fmaxf(rm, mx);
      float wt = __expf(sc - nm);
      float ts = wt;
      for (int off = 1; off < 16; off <<= 1) ts += __shfl_xor(ts, off, 32);
      wts[tid] = wt;
      if (tid == 0) {
        float corr = __expf(rm - nm);
        sCorr   = corr;
        sRunSum = sRunSum * corr + ts;
        sRunMax = nm;
      }
    }
    __syncthreads();                                   // B5

    // ---- pooled accumulator update ----
    {
      float corr = sCorr;
      float wl[8];
      #pragma unroll
      for (int r = 0; r < 8; ++r) wl[r] = wts[r + 8 * hhalf];
      #pragma unroll
      for (int j = 0; j < 4; ++j) {
        float acc = 0.f;
        #pragma unroll
        for (int r = 0; r < 8; ++r) acc += wl[r] * acc2[j][r];
        pool[j] = pool[j] * corr + acc;
      }
    }
  }
  __syncthreads();
  // write unnormalized partials (in local-max frame) + (m, s)
  #pragma unroll
  for (int j = 0; j < 4; ++j) {
    float p = pool[j] + __shfl_xor(pool[j], 16, 32);
    if (hhalf == 0)
      part_acc[(size_t)blk * E_ + w * 64 + j * 16 + nloc] = p;
  }
  if (tid == 0) {
    part_ms[blk * 2 + 0] = sRunMax;
    part_ms[blk * 2 + 1] = sRunSum;
  }
}

// ---------------------------------------------------------------------------
// K1b: merge TSPLIT_ partial softmax accumulators per (b,l).
// pooled = sum_c acc_c * exp(m_c - M) / sum_c s_c * exp(m_c - M)
// ---------------------------------------------------------------------------
__global__ __launch_bounds__(256)
void merge_pool_kernel(const float* __restrict__ part_acc,
                       const float* __restrict__ part_ms,
                       float* __restrict__ pooled) {
  __shared__ float w5[TSPLIT_];
  __shared__ float sInv;
  const int bl  = blockIdx.x;
  const int tid = threadIdx.x;
  if (tid == 0) {
    float M = -3.0e38f;
    for (int cc = 0; cc < TSPLIT_; ++cc)
      M = fmaxf(M, part_ms[(bl * TSPLIT_ + cc) * 2 + 0]);
    float S = 0.f;
    for (int cc = 0; cc < TSPLIT_; ++cc) {
      float e = __expf(part_ms[(bl * TSPLIT_ + cc) * 2 + 0] - M);
      w5[cc] = e;
      S += part_ms[(bl * TSPLIT_ + cc) * 2 + 1] * e;
    }
    sInv = 1.0f / S;
  }
  __syncthreads();
  for (int e = tid; e < E_; e += 256) {
    float s = 0.f;
    #pragma unroll
    for (int cc = 0; cc < TSPLIT_; ++cc)
      s += part_acc[(size_t)(bl * TSPLIT_ + cc) * E_ + e] * w5[cc];
    pooled[(size_t)bl * E_ + e] = s * sInv;
  }
}

// ---------------------------------------------------------------------------
// K2: router + blend + fusion + LayerNorm + classifier. One block per b.
// ---------------------------------------------------------------------------
__global__ __launch_bounds__(256)
void router_head_kernel(const float* __restrict__ pooled, const float* __restrict__ noise,
                        const float* __restrict__ Wr1, const float* __restrict__ br1,
                        const float* __restrict__ Wr2, const float* __restrict__ br2,
                        const float* __restrict__ Wn,  const float* __restrict__ bn,
                        const float* __restrict__ ln_g, const float* __restrict__ ln_b,
                        const float* __restrict__ Wc1, const float* __restrict__ bc1,
                        const float* __restrict__ Wc2, const float* __restrict__ bc2,
                        float* __restrict__ logits_out, float* __restrict__ routing_out,
                        float* __restrict__ fusedn_out) {
  __shared__ float ri[RIN_];
  __shared__ float rh[RH_];
  __shared__ float nacc[L_];
  __shared__ float rw[L_];
  __shared__ float fz[E_];
  __shared__ float fn[E_];
  __shared__ float c1[256];
  __shared__ float sMean, sVar;

  const int tid = threadIdx.x;
  const int b   = blockIdx.x;

  for (int i = tid; i < RIN_; i += 256) ri[i] = pooled[(size_t)b * RIN_ + i];
  if (tid < L_) nacc[tid] = 0.f;
  if (tid == 0) { sMean = 0.f; sVar = 0.f; }
  __syncthreads();

  if (tid < RH_) {
    float s = br1[tid];
    for (int i = 0; i < RIN_; ++i) s += ri[i] * Wr1[(size_t)i * RH_ + tid];
    rh[tid] = fmaxf(s, 0.f);
  }
  // noise-scale dot: partition i across threads, reduce via LDS atomics
  float part[L_];
  #pragma unroll
  for (int n = 0; n < L_; ++n) part[n] = 0.f;
  for (int i = tid; i < RIN_; i += 256) {
    float r = ri[i];
    const float* wrow = Wn + (size_t)i * L_;
    #pragma unroll
    for (int n = 0; n < L_; ++n) part[n] += r * wrow[n];
  }
  #pragma unroll
  for (int n = 0; n < L_; ++n) atomicAdd(&nacc[n], part[n]);
  __syncthreads();

  if (tid < L_) {
    float lr = br2[tid];
    for (int j = 0; j < RH_; ++j) lr += rh[j] * Wr2[j * L_ + tid];
    float xns = nacc[tid] + bn[tid];
    float ns  = xns > 0.f ? xns + log1pf(__expf(-xns)) : log1pf(__expf(xns)); // softplus
    float z   = (lr + noise[b * L_ + tid] * ns) * (1.0f / TEMP_);
    float learned = 1.f / (1.f + __expf(-z));
    float r = ALPHA_ * learned + (1.f - ALPHA_) / (float)L_;
    rw[tid] = r;
    routing_out[b * L_ + tid] = r;
  }
  __syncthreads();

  for (int e = tid; e < E_; e += 256) {
    float s = 0.f;
    #pragma unroll
    for (int l = 0; l < L_; ++l) s += ri[l * E_ + e] * rw[l];
    fz[e] = s;
  }
  __syncthreads();
  {
    float sm = 0.f, sq = 0.f;
    for (int e = tid; e < E_; e += 256) { float v = fz[e]; sm += v; sq += v * v; }
    atomicAdd(&sMean, sm);
    atomicAdd(&sVar, sq);
  }
  __syncthreads();
  float mu  = sMean * (1.0f / E_);
  float var = sVar * (1.0f / E_) - mu * mu;
  float inv = rsqrtf(var + 1e-5f);
  for (int e = tid; e < E_; e += 256) {
    float v = (fz[e] - mu) * inv * ln_g[e] + ln_b[e];
    fn[e] = v;
    fusedn_out[b * E_ + e] = v;
  }
  __syncthreads();
  {
    float s = bc1[tid];
    for (int e = 0; e < E_; ++e) s += fn[e] * Wc1[(size_t)e * 256 + tid];
    c1[tid] = fmaxf(s, 0.f);
  }
  __syncthreads();
  if (tid < NC_) {
    float s = bc2[tid];
    for (int j = 0; j < 256; ++j) s += c1[j] * Wc2[j * NC_ + tid];
    logits_out[b * NC_ + tid] = s;
  }
}

// ---------------------------------------------------------------------------
extern "C" void kernel_launch(void* const* d_in, const int* in_sizes, int n_in,
                              void* d_out, int out_size, void* d_ws, size_t ws_size,
                              hipStream_t stream) {
  const float* h     = (const float*)d_in[0];
  const float* noise = (const float*)d_in[1];
  const float* We1   = (const float*)d_in[2];
  const float* be1   = (const float*)d_in[3];
  const float* We2   = (const float*)d_in[4];
  const float* be2   = (const float*)d_in[5];
  const float* wa    = (const float*)d_in[6];
  const float* ba    = (const float*)d_in[7];
  const float* Wr1   = (const float*)d_in[8];
  const float* br1   = (const float*)d_in[9];
  const float* Wr2   = (const float*)d_in[10];
  const float* br2   = (const float*)d_in[11];
  const float* Wn    = (const float*)d_in[12];
  const float* bn    = (const float*)d_in[13];
  const float* ln_g  = (const float*)d_in[14];
  const float* ln_b  = (const float*)d_in[15];
  const float* Wc1   = (const float*)d_in[16];
  const float* bc1   = (const float*)d_in[17];
  const float* Wc2   = (const float*)d_in[18];
  const float* bc2   = (const float*)d_in[19];

  float* out         = (float*)d_out;
  float* logits_out  = out;                       // [B,NC]   = 32
  float* routing_out = out + B_ * NC_;            // [B,L]    = 384
  float* fusedn_out  = routing_out + B_ * L_;     // [B,E]    = 8192
  float* pooled_out  = fusedn_out + B_ * E_;      // [B,L,E]  = 196608

  _Float16* We1h = (_Float16*)d_ws;                       // 24*1024*512 halves
  _Float16* We2h = We1h + (size_t)L_ * D_ * E_;           // 24*512*512 halves
  float* part_acc = (float*)(We2h + (size_t)L_ * E_ * E_); // [B*L*TSPLIT, E]
  float* part_ms  = part_acc + (size_t)B_ * L_ * TSPLIT_ * E_; // [B*L*TSPLIT, 2]

  convert_weights<<<4096, 256, 0, stream>>>(We1, We2, We1h, We2h);
  expert_pool_kernel<<<B_ * L_ * TSPLIT_, 256, 0, stream>>>(
      h, We1h, We2h, be1, be2, wa, ba, part_acc, part_ms);
  merge_pool_kernel<<<B_ * L_, 256, 0, stream>>>(part_acc, part_ms, pooled_out);
  router_head_kernel<<<B_, 256, 0, stream>>>(pooled_out, noise, Wr1, br1, Wr2, br2,
                                             Wn, bn, ln_g, ln_b, Wc1, bc1, Wc2, bc2,
                                             logits_out, routing_out, fusedn_out);
}
